// DotProductAttention_34299608826646
// MI455X (gfx1250) — matrix operand order repaired and checked by reading, not compiled, not run on hardware
//
#include <hip/hip_runtime.h>
#include <hip/hip_bf16.h>

// Flash-attention for B=16, Lq=Lkv=2048, D=64, fp32 I/O, per-batch valid_len.
//
// Pipeline:
//   1) pre-pass A: K (f32, [b][k][d]) -> Kh (f16, same layout) in workspace
//   2) pre-pass B: V (f32, [b][k][d]) -> VT (f16, [b][d][k]) via LDS transpose
//   3) main: one wave per 16-row q-tile. S^T = Kh * Q^T with
//      v_wmma_f32_16x16x32_f16 (q = lane, key = VGPR) so online-softmax stats
//      are in-lane; O^T = VT * P^T with P^T built from score C-fragments via
//      v_permlanex16 half swaps (no LDS). 1/sqrt(d)*log2(e) folded into the Q
//      conversion, probabilities via raw v_exp_f32. Length mask applied only
//      in the single boundary chunk; uniform scalar early-exit at valid_len.

typedef __attribute__((ext_vector_type(16))) _Float16 v16h;
typedef __attribute__((ext_vector_type(8)))  _Float16 v8h;
typedef __attribute__((ext_vector_type(8)))  float    v8f;

#define WMMA_F16F32(A, Bm, C) \
  __builtin_amdgcn_wmma_f32_16x16x32_f16(false, (A), false, (Bm), (short)0, (C), false, false)

constexpr int   kB    = 16;
constexpr int   kLQ   = 2048;
constexpr int   kLKV  = 2048;
constexpr int   kD    = 64;
constexpr float kLog2e  = 1.44269504088896341f;
constexpr float kQScale = 0.125f * kLog2e;          // 1/sqrt(64) * log2(e)
constexpr float kMask2  = -1000000.0f * kLog2e;     // MASK_VALUE in log2 domain

// xor-16 half swap: pure-VALU v_permlanex16 (lane i <- lane i^16)
__device__ __forceinline__ float xor16(float x) {
#if __has_builtin(__builtin_amdgcn_permlanex16)
  int i = __float_as_int(x);
  i = __builtin_amdgcn_permlanex16(i, i, 0x76543210, 0xfedcba98, false, false);
  return __int_as_float(i);
#else
  return __shfl_xor(x, 16, 32);
#endif
}

__device__ __forceinline__ float fast_exp2(float x) {
#if __has_builtin(__builtin_amdgcn_exp2f)
  return __builtin_amdgcn_exp2f(x);     // raw v_exp_f32; inputs are finite <= 0
#else
  return exp2f(x);
#endif
}

// ---------- pre-pass A: f32 -> f16 elementwise (8 elems/thread) ----------
__global__ __launch_bounds__(256)
void cvt_f16_kernel(const float* __restrict__ src, _Float16* __restrict__ dst, int n8) {
  int i = blockIdx.x * 256 + threadIdx.x;
  if (i >= n8) return;
  const float4* s = (const float4*)src + (size_t)i * 2;
  float4 a = s[0], b = s[1];
  v8h h;
  h[0]=(_Float16)a.x; h[1]=(_Float16)a.y; h[2]=(_Float16)a.z; h[3]=(_Float16)a.w;
  h[4]=(_Float16)b.x; h[5]=(_Float16)b.y; h[6]=(_Float16)b.z; h[7]=(_Float16)b.w;
  *(v8h*)(dst + (size_t)i * 8) = h;
}

// ---------- pre-pass B: V [b][k][d] f32 -> VT [b][d][k] f16 ----------
__global__ __launch_bounds__(256)
void vtrans_kernel(const float* __restrict__ V, _Float16* __restrict__ VT) {
  const int b  = blockIdx.x >> 5;           // 32 slabs of 64 keys per batch
  const int k0 = (blockIdx.x & 31) * 64;
  __shared__ float tile[64][65];            // 64 keys x 64 dims, padded
  const float* src = V + ((size_t)b * kLKV + k0) * kD;
  const int t = threadIdx.x;
  #pragma unroll
  for (int i = 0; i < 4; ++i) {             // 1024 float4 = 64x64 floats
    int idx4 = t + 256 * i;
    int row  = idx4 >> 4;                   // 16 float4 per 64-float row
    int c4   = idx4 & 15;
    float4 f = ((const float4*)src)[idx4];
    tile[row][c4 * 4 + 0] = f.x; tile[row][c4 * 4 + 1] = f.y;
    tile[row][c4 * 4 + 2] = f.z; tile[row][c4 * 4 + 3] = f.w;
  }
  __syncthreads();
  const int d  = t >> 2;                    // 0..63
  const int ks = (t & 3) * 16;              // 16 keys per thread
  v8h h0, h1;
  #pragma unroll
  for (int e = 0; e < 8; ++e) {
    h0[e] = (_Float16)tile[ks + e][d];
    h1[e] = (_Float16)tile[ks + 8 + e][d];
  }
  _Float16* out = VT + (size_t)b * kD * kLKV + (size_t)d * kLKV + k0 + ks;
  *(v8h*)(out)     = h0;
  *(v8h*)(out + 8) = h1;
}

// ---------- main flash-attention kernel ----------
__global__ __launch_bounds__(32)
void fa_wmma_kernel(const float*    __restrict__ Qp,
                    const _Float16* __restrict__ Kh,
                    const _Float16* __restrict__ VT,
                    const int*      __restrict__ vlens,
                    float*          __restrict__ Op)
{
  const int tile  = blockIdx.x;             // 0 .. B*(LQ/16)-1
  const int b     = tile >> 7;              // 128 q-tiles per batch
  const int qbase = (tile & 127) << 4;
  const int lane  = threadIdx.x & 31;
  const int hi    = lane >> 4;
  const int l15   = lane & 15;
  const int vlen  = vlens[b];

  const float*    Qb  = Qp + ((size_t)b * kLQ + qbase) * kD;
  const _Float16* Khb = Kh + (size_t)b * kLKV * kD;
  const _Float16* VTb = VT + (size_t)b * kD * kLKV;

  // Q^T B-fragments (persistent), scale folded in.
  // B layout (32x16): lane n -> col q=n, elems d 0..7,16..23; lane n+16 -> d 8..15,24..31.
  v16h qf[2];
  {
    const float4* qr = (const float4*)(Qb + (size_t)l15 * kD);
    #pragma unroll
    for (int j = 0; j < 2; ++j) {
      #pragma unroll
      for (int q4 = 0; q4 < 4; ++q4) {
        float4 f = qr[j * 8 + hi * 2 + q4 + (q4 >> 1) * 2];
        qf[j][q4 * 4 + 0] = (_Float16)(f.x * kQScale);
        qf[j][q4 * 4 + 1] = (_Float16)(f.y * kQScale);
        qf[j][q4 * 4 + 2] = (_Float16)(f.z * kQScale);
        qf[j][q4 * 4 + 3] = (_Float16)(f.w * kQScale);
      }
    }
  }

  float mrun = -3.0e38f, lrun = 0.0f;
  v8f acc[4] = {v8f{}, v8f{}, v8f{}, v8f{}};

  #pragma unroll 1
  for (int kc = 0; kc < kLKV; kc += 32) {
    if (kc >= vlen) break;                  // uniform per batch -> scalar branch

    if (kc + 32 < kLKV) {                   // hint next chunk into caches
      __builtin_prefetch(Khb + (size_t)(kc + 32 + lane) * kD, 0, 3);
      __builtin_prefetch(VTb + (size_t)(lane * 2) * kLKV + kc + 32, 0, 3);
    }

    // ---- issue ALL loads for this chunk up front so their latency hides
    // under the score WMMAs + softmax.
    // V^T A-fragments: lane -> row d = t*16+l15, elems keys kc+hi*16+{0..15}
    v16h vf[4];
    #pragma unroll
    for (int t = 0; t < 4; ++t) {
      const _Float16* vr = VTb + (size_t)(t * 16 + l15) * kLKV + kc + hi * 16;
      v8h lo  = *(const v8h*)(vr);
      v8h hi8 = *(const v8h*)(vr + 8);
      vf[t] = __builtin_shufflevector(lo, hi8, 0,1,2,3,4,5,6,7,8,9,10,11,12,13,14,15);
    }

    // ---- S^T tiles (log2-scaled scores): T0 = keys kc..+15, T1 = kc+16..+31.
    // C layout: lane = q column, VGPR r = key hi*8 + r within tile.
    v8f st[2];
    #pragma unroll
    for (int t = 0; t < 2; ++t) {
      const int kb = kc + t * 16;
      const _Float16* kr = Khb + (size_t)(kb + l15) * kD;
      v8f s = v8f{};
      #pragma unroll
      for (int j = 0; j < 2; ++j) {
        // A layout (16x32): lane -> key row l15, elems d = 32j+hi*8+{0..7,16..23}
        v8h lo  = *(const v8h*)(kr + j * 32 + hi * 8);
        v8h hi8 = *(const v8h*)(kr + j * 32 + hi * 8 + 16);
        v16h a = __builtin_shufflevector(lo, hi8, 0,1,2,3,4,5,6,7,8,9,10,11,12,13,14,15);
        s = WMMA_F16F32(a, qf[j], s);
      }
      st[t] = s;
    }
    v8f s0 = st[0], s1 = st[1];

    // ---- length mask: only the single boundary chunk needs it (uniform test)
    if (kc + 32 > vlen) {
      #pragma unroll
      for (int r = 0; r < 8; ++r) {
        int key0 = kc + hi * 8 + r;
        int key1 = key0 + 16;
        s0[r] = (key0 < vlen) ? s0[r] : kMask2;
        s1[r] = (key1 < vlen) ? s1[r] : kMask2;
      }
    }

    // ---- online softmax in log2 domain (per q = lane&15; one xor-16 merge)
    float cm = -3.0e38f;
    #pragma unroll
    for (int r = 0; r < 8; ++r) cm = fmaxf(cm, fmaxf(s0[r], s1[r]));
    cm = fmaxf(cm, xor16(cm));
    float mnew  = fmaxf(mrun, cm);
    float alpha = fast_exp2(mrun - mnew);
    float lsum = 0.0f;
    #pragma unroll
    for (int r = 0; r < 8; ++r) {
      s0[r] = fast_exp2(s0[r] - mnew); lsum += s0[r];
      s1[r] = fast_exp2(s1[r] - mnew); lsum += s1[r];
    }
    lsum += xor16(lsum);
    lrun = lrun * alpha + lsum;
    mrun = mnew;
    #pragma unroll
    for (int r = 0; r < 8; ++r) {
      acc[0][r] *= alpha; acc[1][r] *= alpha;
      acc[2][r] *= alpha; acc[3][r] *= alpha;
    }

    // ---- P^T as B-fragment (32 keys x 16 q): xor-16 supplies missing half
    v16h pf;
    #pragma unroll
    for (int r = 0; r < 8; ++r) {
      float o0 = xor16(s0[r]);
      float o1 = xor16(s1[r]);
      pf[r]     = (_Float16)(hi ? o1 : s0[r]);
      pf[r + 8] = (_Float16)(hi ? s1[r] : o0);
    }

    // ---- O^T += VT * P^T : four 16-row d-tiles (fragments preloaded above)
    #pragma unroll
    for (int t = 0; t < 4; ++t) {
      acc[t] = WMMA_F16F32(vf[t], pf, acc[t]);
    }
  }

  // ---- normalize + store: lane holds q = l15, d = t*16 + hi*8 + r
  float inv = 1.0f / lrun;
  float* orow = Op + ((size_t)b * kLQ + qbase + l15) * kD + hi * 8;
  #pragma unroll
  for (int t = 0; t < 4; ++t) {
    float4 lo4, hi4;
    lo4.x = acc[t][0] * inv; lo4.y = acc[t][1] * inv;
    lo4.z = acc[t][2] * inv; lo4.w = acc[t][3] * inv;
    hi4.x = acc[t][4] * inv; hi4.y = acc[t][5] * inv;
    hi4.z = acc[t][6] * inv; hi4.w = acc[t][7] * inv;
    *(float4*)(orow + t * 16)     = lo4;
    *(float4*)(orow + t * 16 + 4) = hi4;
  }
}

extern "C" void kernel_launch(void* const* d_in, const int* in_sizes, int n_in,
                              void* d_out, int out_size, void* d_ws, size_t ws_size,
                              hipStream_t stream) {
  const float* Q  = (const float*)d_in[0];
  const float* K  = (const float*)d_in[1];
  const float* V  = (const float*)d_in[2];
  const int*   vl = (const int*)d_in[3];
  float* out = (float*)d_out;
  (void)in_sizes; (void)n_in; (void)out_size; (void)ws_size;

  const size_t kvElems = (size_t)kB * kLKV * kD;      // 2M elements
  _Float16* Kh = (_Float16*)d_ws;                     // 4 MB
  _Float16* VT = Kh + kvElems;                        // 4 MB

  int n8 = (int)(kvElems / 8);
  cvt_f16_kernel<<<(n8 + 255) / 256, 256, 0, stream>>>(K, Kh, n8);
  vtrans_kernel<<<kB * (kLKV / 64), 256, 0, stream>>>(V, VT);
  fa_wmma_kernel<<<kB * (kLQ / 16), 32, 0, stream>>>(Q, Kh, VT, vl, out);
}
